// FWFormer_88845693485168
// MI455X (gfx1250) — compile-verified, matrix operand order checked by reading, not run on hardware
//
#include <hip/hip_runtime.h>
#include <hip/hip_bf16.h>

typedef __attribute__((ext_vector_type(16))) _Float16     v16h;
typedef __attribute__((ext_vector_type(8)))  float        v8f;
typedef __attribute__((ext_vector_type(4)))  unsigned int u32x4;
typedef __attribute__((ext_vector_type(8)))  unsigned int u32x8;

#define NB    2
#define HH    64
#define WW    64
#define CC    256
#define NHEAD 8
#define DH    32
#define LL    (HH * WW)          // 4096
#define MROWS (NB * LL)          // 8192

// ---------------------------------------------------------------------------
// 16-bit A-matrix 16x32 fragment K-index map (CDNA5 ISA §7.12.2):
//   lanes 0-15 : VGPR0..3 hold K=0..7 pairs, VGPR4..7 hold K=16..23 pairs
//   lanes 16-31: +8 on the K base
__device__ __forceinline__ int kmap(int h, int grp) {
    int v    = h >> 1;
    int base = (v < 4) ? (2 * v) : (16 + 2 * (v - 4));
    return base + grp * 8 + (h & 1);
}

__device__ __forceinline__ v8f wmma_f16(v16h a, v16h b, v8f c) {
    return __builtin_amdgcn_wmma_f32_16x16x32_f16(
        /*neg_a=*/false, a, /*neg_b=*/false, b,
        /*c_mod=*/(short)0, c, /*reuse_a=*/false, /*reuse_b=*/false);
}

// ---------------------------------------------------------------------------
// Tensor Data Mover: DMA a 128-row x 32-col f32 tile (row stride 4096 elems)
// from global memory into LDS. 2D tensor -> D# groups 0 and 1 only
// (VADDR2/VADDR3 omitted == NULL per ISA). Tracked with TENSORcnt.
__device__ __forceinline__ void tdm_load_mask_tile(const float* gsrc,
                                                   unsigned lds_off) {
    unsigned long long ga = (unsigned long long)(uintptr_t)gsrc;
    u32x4 g0;
    g0.x = 1u;                                   // count = 1 valid descriptor
    g0.y = lds_off;                              // lds_addr (bytes)
    g0.z = (unsigned)ga;                         // global_addr[31:0]
    g0.w = ((unsigned)(ga >> 32) & 0x01FFFFFFu)  // global_addr[56:32]
         | (2u << 30);                           // type = 2 ("image")
    u32x8 g1;
    g1[0] = 2u << 16;        // data_size = 2 (4 bytes); no pad/iterate/mcast
    g1[1] = 4096u << 16;     // tensor_dim0 = 4096 (bits [79:48], low half)
    g1[2] = 4096u << 16;     // tensor_dim0 hi=0 | tensor_dim1 = 4096 (low half)
    g1[3] = 32u << 16;       // tensor_dim1 hi=0 | tile_dim0 = 32
    g1[4] = 128u;            // tile_dim1 = 128, tile_dim2 = 0
    g1[5] = 4096u;           // tensor_dim0_stride[31:0] = 4096 elems
    g1[6] = 0u;              // stride hi, tensor_dim1_stride lo (unused, 2D)
    g1[7] = 0u;
    asm volatile("tensor_load_to_lds %0, %1" :: "s"(g0), "s"(g1) : "memory");
}

// ---------------------------------------------------------------------------
// QKV projection: Q = xWq^T + bq ; K = (xWk^T + bk) * d^-0.5 ; V = xWv^T + bv
__global__ void qkv_kernel(const float* __restrict__ x,
                           const float* __restrict__ Wq, const float* __restrict__ bq,
                           const float* __restrict__ Wk, const float* __restrict__ bk,
                           const float* __restrict__ Wv, const float* __restrict__ bv,
                           _Float16* __restrict__ q16, _Float16* __restrict__ k16,
                           _Float16* __restrict__ v16, float* __restrict__ v32) {
    const int lane = threadIdx.x & 31;
    const int wave = threadIdx.x >> 5;
    const int grp  = lane >> 4;
    const int lm   = lane & 15;

    const int tileId = blockIdx.x * 8 + wave;       // 8192 tiles total
    const int mtile  = tileId >> 4;
    const int ntile  = tileId & 15;
    const int row0   = mtile * 16;
    const int col0   = ntile * 16;

    v8f accQ = {}, accK = {}, accV = {};

    for (int kk = 0; kk < CC; kk += 32) {
        v16h aX;
#pragma unroll
        for (int h = 0; h < 16; ++h)
            aX[h] = (_Float16)x[(size_t)(row0 + lm) * CC + kk + kmap(h, grp)];
        v16h bQ, bK, bV;
#pragma unroll
        for (int h = 0; h < 16; ++h) {
            int k = kk + kmap(h, grp);
            size_t wi = (size_t)(col0 + lm) * CC + k;
            bQ[h] = (_Float16)Wq[wi];
            bK[h] = (_Float16)Wk[wi];
            bV[h] = (_Float16)Wv[wi];
        }
        accQ = wmma_f16(aX, bQ, accQ);
        accK = wmma_f16(aX, bK, accK);
        accV = wmma_f16(aX, bV, accV);
    }

    const float scaling = 0.17677669529663687f;     // 32^-0.5
#pragma unroll
    for (int j = 0; j < 8; ++j) {
        int r = row0 + j + 8 * grp;
        int n = col0 + lm;
        size_t o = (size_t)r * CC + n;
        float qv = accQ[j] + bq[n];
        float kv = (accK[j] + bk[n]) * scaling;
        float vv = accV[j] + bv[n];
        q16[o] = (_Float16)qv;
        k16[o] = (_Float16)kv;
        v16[o] = (_Float16)vv;
        v32[o] = vv;
    }
}

// ---------------------------------------------------------------------------
// 5x5 depthwise conv, SAME zero padding. kern layout (ky,kx,1,c).
__global__ void lepe_kernel(const float* __restrict__ v32,
                            const float* __restrict__ kern,
                            const float* __restrict__ bias,
                            float* __restrict__ lepe32) {
    int idx = blockIdx.x * blockDim.x + threadIdx.x;   // B*H*W*C = 2^21
    int c  = idx & 255;
    int w  = (idx >> 8) & 63;
    int h  = (idx >> 14) & 63;
    int b  = idx >> 20;
    float acc = bias[c];
#pragma unroll
    for (int dy = -2; dy <= 2; ++dy) {
#pragma unroll
        for (int dx = -2; dx <= 2; ++dx) {
            int hh = h + dy, ww = w + dx;
            if (hh >= 0 && hh < HH && ww >= 0 && ww < WW) {
                float xin = v32[(((size_t)b * HH + hh) * WW + ww) * CC + c];
                float kv  = kern[((size_t)(dy + 2) * 5 + (dx + 2)) * CC + c];
                acc = fmaf(xin, kv, acc);
            }
        }
    }
    lepe32[idx] = acc;
}

// ---------------------------------------------------------------------------
// Flash-attention with additive decay mask.
// Block = 128 query rows of one (b, head); 8 waves x 16-row strips.
// Mask tiles (128x32 f32) are DMA'd into LDS by the TDM, double-buffered;
// batch index is the fastest-varying block dim so the two batches that share
// a mask slab run back-to-back and hit in the 192 MB L2.
__global__ void attn_kernel(const _Float16* __restrict__ q16,
                            const _Float16* __restrict__ k16,
                            const _Float16* __restrict__ v16,
                            const float* __restrict__ mask,
                            float* __restrict__ attn32) {
    __shared__ float    mask_lds[2][128 * 32];      // 2 x 16 KB TDM buffers
    __shared__ _Float16 lds_p[8][16 * 32];          // per-wave P staging (8 KB)

    const int lane = threadIdx.x & 31;
    const int wave = threadIdx.x >> 5;
    const int grp  = lane >> 4;
    const int lm   = lane & 15;

    const int b     = blockIdx.x & 1;               // fastest: pair L2 reuse
    const int head  = (blockIdx.x >> 1) & 7;
    const int mtile = blockIdx.x >> 4;              // 32 tiles of 128 rows
    const int rowblk  = mtile * 128;
    const int rowbase = rowblk + wave * 16;
    const int hc = head * DH;

    const float* mask_blk = mask + ((size_t)head * LL + rowblk) * LL;

    // Q fragment (16 rows x 32 head-dims), resident for the whole pass
    v16h aQ;
#pragma unroll
    for (int h = 0; h < 16; ++h)
        aQ[h] = q16[(size_t)(b * LL + rowbase + lm) * CC + hc + kmap(h, grp)];

    v8f o0 = {}, o1 = {};
    float mrow[8], lrow[8];
#pragma unroll
    for (int j = 0; j < 8; ++j) { mrow[j] = -1e30f; lrow[j] = 0.0f; }

    // TDM prologue: stage first mask tile
    if (wave == 0)
        tdm_load_mask_tile(mask_blk,
                           (unsigned)(uintptr_t)(void*)&mask_lds[0][0]);

    const int NT = LL / 32;                         // 128 key tiles
    for (int t = 0; t < NT; ++t) {
        const int kt  = t * 32;
        const int cur = t & 1;

        if (wave == 0)
            __builtin_amdgcn_s_wait_tensorcnt(0);   // tile t landed in LDS
        __syncthreads();                            // publish to all waves
        if (wave == 0 && t + 1 < NT)                // buf[cur^1]'s readers are
            tdm_load_mask_tile(                     // past the barrier above
                mask_blk + kt + 32,
                (unsigned)(uintptr_t)(void*)&mask_lds[cur ^ 1][0]);

        // --- S = Q @ K^T : B[k=dvec][n=key] = K[key][dvec] ---
        v16h bK0, bK1;
#pragma unroll
        for (int h = 0; h < 16; ++h) {
            int dv = hc + kmap(h, grp);
            bK0[h] = k16[(size_t)(b * LL + kt      + lm) * CC + dv];
            bK1[h] = k16[(size_t)(b * LL + kt + 16 + lm) * CC + dv];
        }
        v8f z = {};
        v8f S0 = wmma_f16(aQ, bK0, z);
        v8f S1 = wmma_f16(aQ, bK1, z);

        // --- add decay mask from the LDS-staged tile ---
#pragma unroll
        for (int j = 0; j < 8; ++j) {
            int r = wave * 16 + j + 8 * grp;        // row within 128-row tile
            S0[j] += mask_lds[cur][r * 32 + lm];
            S1[j] += mask_lds[cur][r * 32 + 16 + lm];
        }

        // --- online softmax (row = 16 lanes of a half-wave group) ---
#pragma unroll
        for (int j = 0; j < 8; ++j) {
            float tm = fmaxf(S0[j], S1[j]);
#pragma unroll
            for (int off = 8; off >= 1; off >>= 1)
                tm = fmaxf(tm, __shfl_xor(tm, off, 32));
            float newm  = fmaxf(mrow[j], tm);
            float scale = __expf(mrow[j] - newm);
            float p0 = __expf(S0[j] - newm);
            float p1 = __expf(S1[j] - newm);
            S0[j] = p0; S1[j] = p1;
            float rs = p0 + p1;
#pragma unroll
            for (int off = 8; off >= 1; off >>= 1)
                rs += __shfl_xor(rs, off, 32);
            lrow[j] = lrow[j] * scale + rs;
            mrow[j] = newm;
            o0[j] *= scale;
            o1[j] *= scale;
        }

        // --- C-layout P (f32) -> A-layout P (f16) via per-wave LDS ---
#pragma unroll
        for (int j = 0; j < 8; ++j) {
            int r = j + 8 * grp;
            lds_p[wave][r * 32 + lm]      = (_Float16)S0[j];
            lds_p[wave][r * 32 + 16 + lm] = (_Float16)S1[j];
        }
        asm volatile("s_wait_dscnt 0" ::: "memory");   // intra-wave LDS RAW
        v16h aP;
#pragma unroll
        for (int h = 0; h < 16; ++h)
            aP[h] = lds_p[wave][lm * 32 + kmap(h, grp)];

        // --- O += P @ V : B[k=keyoff][n=dcol] = V[kt+keyoff][dcol] ---
        v16h bV0, bV1;
#pragma unroll
        for (int h = 0; h < 16; ++h) {
            int ko = kmap(h, grp);
            bV0[h] = v16[(size_t)(b * LL + kt + ko) * CC + hc + lm];
            bV1[h] = v16[(size_t)(b * LL + kt + ko) * CC + hc + 16 + lm];
        }
        o0 = wmma_f16(aP, bV0, o0);
        o1 = wmma_f16(aP, bV1, o1);
    }

#pragma unroll
    for (int j = 0; j < 8; ++j) {
        int r = rowbase + j + 8 * grp;
        float inv = 1.0f / lrow[j];
        size_t o = (size_t)(b * LL + r) * CC + hc;
        attn32[o + lm]      = o0[j] * inv;
        attn32[o + 16 + lm] = o1[j] * inv;
    }
}

// ---------------------------------------------------------------------------
// Output projection: out = (attn + lepe) @ Wo^T + bo  (fp32 output)
__global__ void proj_kernel(const float* __restrict__ attn32,
                            const float* __restrict__ lepe32,
                            const float* __restrict__ Wo,
                            const float* __restrict__ bo,
                            float* __restrict__ out) {
    const int lane = threadIdx.x & 31;
    const int wave = threadIdx.x >> 5;
    const int grp  = lane >> 4;
    const int lm   = lane & 15;

    const int tileId = blockIdx.x * 8 + wave;
    const int mtile  = tileId >> 4;
    const int ntile  = tileId & 15;
    const int row0   = mtile * 16;
    const int col0   = ntile * 16;

    v8f acc = {};
    for (int kk = 0; kk < CC; kk += 32) {
        v16h aA;
#pragma unroll
        for (int h = 0; h < 16; ++h) {
            size_t i = (size_t)(row0 + lm) * CC + kk + kmap(h, grp);
            aA[h] = (_Float16)(attn32[i] + lepe32[i]);
        }
        v16h bW;
#pragma unroll
        for (int h = 0; h < 16; ++h)
            bW[h] = (_Float16)Wo[(size_t)(col0 + lm) * CC + kk + kmap(h, grp)];
        acc = wmma_f16(aA, bW, acc);
    }
#pragma unroll
    for (int j = 0; j < 8; ++j) {
        int r = row0 + j + 8 * grp;
        int n = col0 + lm;
        out[(size_t)r * CC + n] = acc[j] + bo[n];
    }
}

// ---------------------------------------------------------------------------
extern "C" void kernel_launch(void* const* d_in, const int* in_sizes, int n_in,
                              void* d_out, int out_size, void* d_ws, size_t ws_size,
                              hipStream_t stream) {
    const float* x      = (const float*)d_in[0];
    const float* mask   = (const float*)d_in[1];
    const float* Wq     = (const float*)d_in[2];
    const float* bq     = (const float*)d_in[3];
    const float* Wk     = (const float*)d_in[4];
    const float* bk     = (const float*)d_in[5];
    const float* Wv     = (const float*)d_in[6];
    const float* bv     = (const float*)d_in[7];
    const float* lepe_w = (const float*)d_in[8];
    const float* lepe_b = (const float*)d_in[9];
    const float* Wo     = (const float*)d_in[10];
    const float* bo     = (const float*)d_in[11];
    float* out = (float*)d_out;

    // Workspace layout (MROWS*CC = 2,097,152 elements)
    char* ws = (char*)d_ws;
    const size_t NE = (size_t)MROWS * CC;
    _Float16* q16  = (_Float16*)(ws);                      //  4 MB
    _Float16* k16  = (_Float16*)(ws + 2 * NE);             //  4 MB
    _Float16* v16  = (_Float16*)(ws + 4 * NE);             //  4 MB
    float* v32     = (float*)   (ws + 6 * NE);             //  8 MB
    float* lepe32  = (float*)   (ws + 6 * NE + 4 * NE);    //  8 MB
    float* attn32  = (float*)   (ws + 6 * NE + 8 * NE);    //  8 MB

    qkv_kernel<<<1024, 256, 0, stream>>>(x, Wq, bq, Wk, bk, Wv, bv,
                                         q16, k16, v16, v32);
    lepe_kernel<<<8192, 256, 0, stream>>>(v32, lepe_w, lepe_b, lepe32);
    attn_kernel<<<512, 256, 0, stream>>>(q16, k16, v16, mask, attn32);
    proj_kernel<<<1024, 256, 0, stream>>>(attn32, lepe32, Wo, bo, out);
}
